// HQSc_DMPNN_56453050139301
// MI455X (gfx1250) — matrix operand order repaired and checked by reading, not compiled, not run on hardware
//
#include <hip/hip_runtime.h>
#include <hip/hip_bf16.h>

#define N_NODES  100000
#define N_EDGES  1000000
#define D_IN     64
#define D_EDGE   16
#define HDIM     64
#define N_QUBITS 8
#define N_GRAPHS 1024
#define STEPS    4
#define LN_EPS   1e-5f

typedef __attribute__((ext_vector_type(16))) _Float16 v16h;
typedef __attribute__((ext_vector_type(8)))  _Float16 v8h;
typedef __attribute__((ext_vector_type(2)))  _Float16 v2h;
typedef __attribute__((ext_vector_type(8)))  float    v8f;

// ---- WMMA helpers -----------------------------------------------------------
// A-frag (16x32 f16, CDNA5 layout): lane l holds row M=l&15, k-half kh=l>>4.
//   VGPR v(0..3): K = kh*8 + 2v..2v+1 ; v(4..7): K = 16 + kh*8 + ...
__device__ __forceinline__ v16h ldsA(const _Float16* tile, int strideH,
                                     int M, int kh, int chunk) {
  const _Float16* p = tile + M * strideH + chunk * 32 + kh * 8;
  v8h lo = *(const v8h*)p;
  v8h hi = *(const v8h*)(p + 16);
  v16h a;
#pragma unroll
  for (int i = 0; i < 8; ++i) { a[i] = lo[i]; a[i + 8] = hi[i]; }
  return a;
}
// B-frag (32x16 f16): lane l holds column n, kh=l>>4; VGPR v(0..3): K = kh*8 + 2v..2v+1 ; v(4..7): K = 16 + kh*8 + ...
// => 8 f16 at K = chunk*32 + kh*8 and 8 at +16 in a [n][K] transposed weight tile.
__device__ __forceinline__ v16h ldsB(const _Float16* wT, int strideH,
                                     int n, int kh, int chunk) {
  const _Float16* p = wT + n * strideH + chunk * 32 + kh * 8;
  v8h lo = *(const v8h*)p;
  v8h hi = *(const v8h*)(p + 16);
  v16h b;
#pragma unroll
  for (int i = 0; i < 8; ++i) { b[i] = lo[i]; b[i + 8] = hi[i]; }
  return b;
}
__device__ __forceinline__ v8f wmma_f16(v16h a, v16h b, v8f c) {
  return __builtin_amdgcn_wmma_f32_16x16x32_f16(false, a, false, b,
                                                (short)0, c, false, false);
}
__device__ __forceinline__ float frelu(float v) { return v > 0.f ? v : 0.f; }

// ---- h0 = relu([x[src] | ef] @ wi + bi); fused scatter-add into nodeA -------
__global__ void __launch_bounds__(128) k_edge_init(
    const float* __restrict__ x, const float* __restrict__ ef,
    const int* __restrict__ src, const int* __restrict__ dst,
    const float* __restrict__ wi, const float* __restrict__ bi,
    float* __restrict__ h0, float* __restrict__ nodeNext) {
  __shared__ __align__(16) _Float16 sWT[64 * 96];      // wi^T, K padded 80->96
  __shared__ __align__(16) _Float16 sT[4][16 * 96];
  const int tid = threadIdx.x;
  for (int idx = tid; idx < 64 * 96; idx += 128) {
    int n = idx / 96, k = idx % 96;
    sWT[idx] = (_Float16)((k < 80) ? wi[k * 64 + n] : 0.0f);
  }
  __syncthreads();
  const int lane = tid & 31, wid = tid >> 5;
  const int M = lane & 15, kh = lane >> 4;
  const int tile = blockIdx.x * 4 + wid;
  if (tile >= N_EDGES / 16) return;
  const unsigned base = (unsigned)tile * 16u;
  _Float16* it = sT[wid];
  for (int i = 0; i < 16; ++i) {
    unsigned e = base + i;
    unsigned s = (unsigned)src[e];
#pragma unroll
    for (int t = 0; t < 3; ++t) {
      unsigned c = (unsigned)lane + t * 32u;
      float v = 0.0f;
      if (c < 64u)      v = x[s * 64u + c];
      else if (c < 80u) v = ef[e * 16u + (c - 64u)];
      it[i * 96 + c] = (_Float16)v;
    }
  }
  unsigned sidx[8];
#pragma unroll
  for (int r = 0; r < 8; ++r) sidx[r] = (unsigned)dst[base + r + 8 * kh];
  v16h a0 = ldsA(it, 96, M, kh, 0);
  v16h a1 = ldsA(it, 96, M, kh, 1);
  v16h a2 = ldsA(it, 96, M, kh, 2);
#pragma unroll
  for (int j = 0; j < 4; ++j) {
    int col = j * 16 + M;
    v8f c = {};
    c = wmma_f16(a0, ldsB(sWT, 96, col, kh, 0), c);
    c = wmma_f16(a1, ldsB(sWT, 96, col, kh, 1), c);
    c = wmma_f16(a2, ldsB(sWT, 96, col, kh, 2), c);
    float bv = bi[col];
#pragma unroll
    for (int r = 0; r < 8; ++r) {
      unsigned e = base + r + 8u * kh;
      float v = frelu(c[r] + bv);
      h0[e * 64u + (unsigned)col] = v;
      atomicAdd(nodeNext + (sidx[r] * 64u + (unsigned)col), v); // segsum(dst)
    }
  }
}

// ---- h = relu(h0 + relu((hnode[dst]-hin) @ w1 + b1) @ w2 + b2)
//      + fused scatter-add of the new h into nodeNext via scat index ----------
__global__ void __launch_bounds__(128) k_edge_step(
    const float* __restrict__ hin, float* __restrict__ hout,
    const float* __restrict__ h0,
    const float* __restrict__ hnode, const int* __restrict__ dst,
    const int* __restrict__ scat, float* __restrict__ nodeNext,
    const float* __restrict__ w1, const float* __restrict__ b1,
    const float* __restrict__ w2, const float* __restrict__ b2) {
  __shared__ __align__(16) _Float16 sW1[64 * 64];
  __shared__ __align__(16) _Float16 sW2[64 * 64];
  __shared__ __align__(16) _Float16 sT[4][16 * 64];
  const int tid = threadIdx.x;
  for (int idx = tid; idx < 64 * 64; idx += 128) {   // transpose to [n][k]
    int k = idx >> 6, n = idx & 63;
    sW1[n * 64 + k] = (_Float16)w1[idx];
    sW2[n * 64 + k] = (_Float16)w2[idx];
  }
  __syncthreads();
  const int lane = tid & 31, wid = tid >> 5;
  const int M = lane & 15, kh = lane >> 4;
  const int tile = blockIdx.x * 4 + wid;
  if (tile >= N_EDGES / 16) return;
  const unsigned base = (unsigned)tile * 16u;
  _Float16* mt = sT[wid];
  for (int i = 0; i < 16; ++i) {                     // stage m-tile (fp32->f16)
    unsigned e = base + i;
    unsigned d = (unsigned)dst[e];
    float2 hv = ((const float2*)hin)[e * 32u + (unsigned)lane];
    float2 nv = ((const float2*)hnode)[d * 32u + (unsigned)lane];
    v2h p;
    p[0] = (_Float16)(nv.x - hv.x);
    p[1] = (_Float16)(nv.y - hv.y);
    *(v2h*)(mt + i * 64 + lane * 2) = p;
    __builtin_prefetch(h0 + e * 64u, 0, 1);          // global_prefetch for epilogue
  }
  unsigned sidx[8];
#pragma unroll
  for (int r = 0; r < 8; ++r) sidx[r] = (unsigned)scat[base + r + 8 * kh];
  v16h a0 = ldsA(mt, 64, M, kh, 0);
  v16h a1 = ldsA(mt, 64, M, kh, 1);
#pragma unroll
  for (int j = 0; j < 4; ++j) {                      // t = relu(m@w1+b1)
    int col = j * 16 + M;
    v8f c = {};
    c = wmma_f16(a0, ldsB(sW1, 64, col, kh, 0), c);
    c = wmma_f16(a1, ldsB(sW1, 64, col, kh, 1), c);
    float bv = b1[col];
#pragma unroll
    for (int r = 0; r < 8; ++r)                      // D-layout -> LDS (A relayout)
      mt[(r + 8 * kh) * 64 + col] = (_Float16)frelu(c[r] + bv);
  }
  v16h t0 = ldsA(mt, 64, M, kh, 0);
  v16h t1 = ldsA(mt, 64, M, kh, 1);
#pragma unroll
  for (int j = 0; j < 4; ++j) {                      // h = relu(h0 + t@w2 + b2)
    int col = j * 16 + M;
    v8f c = {};
    c = wmma_f16(t0, ldsB(sW2, 64, col, kh, 0), c);
    c = wmma_f16(t1, ldsB(sW2, 64, col, kh, 1), c);
    float bv = b2[col];
#pragma unroll
    for (int r = 0; r < 8; ++r) {
      unsigned e = base + r + 8u * kh;
      unsigned o = e * 64u + (unsigned)col;          // 32-bit offsets -> saddr+voffset
      float v = frelu(h0[o] + c[r] + bv);
      hout[o] = v;
      atomicAdd(nodeNext + (sidx[r] * 64u + (unsigned)col), v); // next segsum
    }
  }
}

// ---- relu(LN(relu([x | m_node] @ wf + bf))) fused with global mean pool -----
__global__ void __launch_bounds__(128) k_node_final(
    const float* __restrict__ x, const float* __restrict__ mnode,
    const float* __restrict__ wf, const float* __restrict__ bf,
    const float* __restrict__ lng, const float* __restrict__ lnb,
    const int* __restrict__ batch,
    float* __restrict__ sums, float* __restrict__ cnts) {
  __shared__ __align__(16) _Float16 sWT[64 * 128];
  __shared__ __align__(16) _Float16 sT[4][16 * 128];
  __shared__ __align__(16) float    sR[4][16 * 64];
  const int tid = threadIdx.x;
  for (int idx = tid; idx < 64 * 128; idx += 128) {
    int k = idx >> 6, n = idx & 63;
    sWT[n * 128 + k] = (_Float16)wf[idx];
  }
  __syncthreads();
  const int lane = tid & 31, wid = tid >> 5;
  const int M = lane & 15, kh = lane >> 4;
  const int tile = blockIdx.x * 4 + wid;
  if (tile >= N_NODES / 16) return;
  const unsigned base = (unsigned)tile * 16u;
  _Float16* it = sT[wid];
  float* rt = sR[wid];
  for (int i = 0; i < 16; ++i) {
    unsigned nd = base + i;
#pragma unroll
    for (int t = 0; t < 4; ++t) {
      unsigned c = (unsigned)lane + t * 32u;
      float v = (c < 64u) ? x[nd * 64u + c] : mnode[nd * 64u + (c - 64u)];
      it[i * 128 + c] = (_Float16)v;
    }
  }
  v16h a0 = ldsA(it, 128, M, kh, 0), a1 = ldsA(it, 128, M, kh, 1);
  v16h a2 = ldsA(it, 128, M, kh, 2), a3 = ldsA(it, 128, M, kh, 3);
#pragma unroll
  for (int j = 0; j < 4; ++j) {
    int col = j * 16 + M;
    v8f c = {};
    c = wmma_f16(a0, ldsB(sWT, 128, col, kh, 0), c);
    c = wmma_f16(a1, ldsB(sWT, 128, col, kh, 1), c);
    c = wmma_f16(a2, ldsB(sWT, 128, col, kh, 2), c);
    c = wmma_f16(a3, ldsB(sWT, 128, col, kh, 3), c);
    float bv = bf[col];
#pragma unroll
    for (int r = 0; r < 8; ++r)
      rt[(r + 8 * kh) * 64 + col] = frelu(c[r] + bv);
  }
  if (lane < 16) {            // LayerNorm + ReLU + mean-pool scatter, row/lane
    int row = lane;
    float s = 0.f, ss = 0.f;
    for (int c2 = 0; c2 < 64; ++c2) { float v = rt[row * 64 + c2]; s += v; ss += v * v; }
    float mu = s * (1.f / 64.f);
    float var = ss * (1.f / 64.f) - mu * mu;
    float rstd = rsqrtf(var + LN_EPS);
    unsigned nd = base + (unsigned)row;
    unsigned g = (unsigned)batch[nd];
    atomicAdd(cnts + g, 1.0f);
    for (int c2 = 0; c2 < 64; ++c2) {
      float v = (rt[row * 64 + c2] - mu) * rstd * lng[c2] + lnb[c2];
      atomicAdd(sums + (g * 64u + (unsigned)c2), frelu(v));
    }
  }
}

// ---- utility kernels --------------------------------------------------------
__global__ void k_zero(float* __restrict__ p, long long n) {
  long long t = (long long)blockIdx.x * blockDim.x + threadIdx.x;
  if (t < n) p[t] = 0.f;
}
__global__ void k_out(const float* __restrict__ sums, const float* __restrict__ cnts,
                      const float* __restrict__ wq, float* __restrict__ out) {
  int t = blockIdx.x * blockDim.x + threadIdx.x;
  if (t >= N_GRAPHS * N_QUBITS) return;
  unsigned g = (unsigned)t >> 3, q = (unsigned)t & 7u;
  float acc = 0.f;
  for (unsigned c = 0; c < 64u; ++c) acc += sums[g * 64u + c] * wq[c * 8u + q];
  float cnt = cnts[g];
  out[t] = acc / (cnt > 1.f ? cnt : 1.f);
}

extern "C" void kernel_launch(void* const* d_in, const int* in_sizes, int n_in,
                              void* d_out, int out_size, void* d_ws, size_t ws_size,
                              hipStream_t stream) {
  const float* x     = (const float*)d_in[0];
  const float* ef    = (const float*)d_in[1];
  const int*   ei    = (const int*)d_in[2];
  const int*   batch = (const int*)d_in[3];
  const float* wi = (const float*)d_in[4];
  const float* bi = (const float*)d_in[5];
  const float* w1 = (const float*)d_in[6];
  const float* b1 = (const float*)d_in[7];
  const float* w2 = (const float*)d_in[8];
  const float* b2 = (const float*)d_in[9];
  const float* wf = (const float*)d_in[10];
  const float* bf = (const float*)d_in[11];
  const float* lng = (const float*)d_in[12];
  const float* lnb = (const float*)d_in[13];
  const float* wq  = (const float*)d_in[14];
  const int* src = ei;
  const int* dst = ei + N_EDGES;

  float* ws    = (float*)d_ws;
  float* h0    = ws;
  float* h     = h0 + (size_t)N_EDGES * HDIM;
  float* nodeA = h + (size_t)N_EDGES * HDIM;
  float* nodeB = nodeA + (size_t)N_NODES * HDIM;
  float* sums  = nodeB + (size_t)N_NODES * HDIM;
  float* cnts  = sums + (size_t)N_GRAPHS * HDIM;
  float* out   = (float*)d_out;

  const int ebGrid = (N_EDGES / 16) / 4;                     // 15625 blocks
  const long long NH = (long long)N_NODES * HDIM;
  const int zNH = (int)((NH + 255) / 256);

  // edge init: h0 + scatter-add(dst) into nodeA
  k_zero<<<zNH, 256, 0, stream>>>(nodeA, NH);
  k_edge_init<<<ebGrid, 128, 0, stream>>>(x, ef, src, dst, wi, bi, h0, nodeA);

  float* ping = nodeA;  // holds segment_sum of current h via dst
  float* pong = nodeB;
  for (int s = 0; s < STEPS; ++s) {
    k_zero<<<zNH, 256, 0, stream>>>(pong, NH);
    const float* hin = (s == 0) ? h0 : h;
    const int* scat = (s == STEPS - 1) ? src : dst;  // last step builds m_node
    k_edge_step<<<ebGrid, 128, 0, stream>>>(hin, h, h0, ping, dst, scat, pong,
                                            w1, b1, w2, b2);
    float* tmp = ping; ping = pong; pong = tmp;
  }
  // ping now holds m_node = segment_sum(h, src)

  k_zero<<<(N_GRAPHS * HDIM + N_GRAPHS + 255) / 256, 256, 0, stream>>>(
      sums, N_GRAPHS * HDIM + N_GRAPHS);                     // sums + cnts contiguous
  const int nbGrid = (N_NODES / 16 + 3) / 4;                 // 1563 blocks
  k_node_final<<<nbGrid, 128, 0, stream>>>(x, ping, wf, bf, lng, lnb, batch,
                                           sums, cnts);
  k_out<<<(N_GRAPHS * N_QUBITS + 63) / 64, 64, 0, stream>>>(sums, cnts, wq, out);
}
